// _Module_5592047419397
// MI455X (gfx1250) — matrix operand order repaired and checked by reading, not compiled, not run on hardware
//
#include <hip/hip_runtime.h>
#include <hip/hip_bf16.h>
#include <math.h>

// ---------------------------------------------------------------------------
// ViT forward (N=8, 384x384x3, patch16 -> L=576, W=768, depth=4, H=12, HD=64,
// MLP=3072). All GEMMs on v_wmma_f32_16x16x32_f16 (f32 accumulate).
// Weights converted+transposed to f16 once; activations produced in f16 by
// their producers so GEMM/attention inner loops have zero conversions.
// Tile staging uses GLOBAL_LOAD_ASYNC_TO_LDS_B128 (ASYNCcnt) when available.
// ---------------------------------------------------------------------------

typedef __attribute__((ext_vector_type(16))) _Float16 v16h;
typedef __attribute__((ext_vector_type(8)))  _Float16 v8h;
typedef __attribute__((ext_vector_type(8)))  float    v8f;
typedef __attribute__((ext_vector_type(4)))  float    v4f;
typedef __attribute__((ext_vector_type(4)))  int      v4i;

#define NIMG   8
#define GRIDP  24           // 384/16 patches per side
#define LTOK   576          // GRIDP*GRIDP
#define MTOK   (NIMG*LTOK)  // 4608 rows
#define WID    768
#define NHEAD  12
#define HDIM   64
#define MLPD   3072
#define DEPTH  4

#if __has_builtin(__builtin_amdgcn_global_load_async_to_lds_b128) && \
    __has_builtin(__builtin_amdgcn_s_wait_asynccnt)
#define HAVE_ASYNC_LDS 1
#endif

#if defined(HAVE_ASYNC_LDS)
typedef __attribute__((address_space(1))) v4i gv4i_t;   // global int4
typedef __attribute__((address_space(3))) v4i lv4i_t;   // LDS int4
#endif

// 16-byte global -> LDS copy (async path uses ASYNCcnt datapath)
__device__ __forceinline__ void copy16_g2l(const _Float16* g, _Float16* l) {
#if defined(HAVE_ASYNC_LDS)
    __builtin_amdgcn_global_load_async_to_lds_b128(
        (gv4i_t*)(const void*)g, (lv4i_t*)(void*)l, 0, 0);
#else
    *(v8h*)l = *(const v8h*)g;
#endif
}
__device__ __forceinline__ void wait_copies() {
#if defined(HAVE_ASYNC_LDS)
    __builtin_amdgcn_s_wait_asynccnt(0);
#endif
}

__device__ __forceinline__ v8f wmma16(v16h a, v16h b, v8f c) {
    // D = A(16x32, f16) x B(32x16, f16) + C(16x16, f32)
    return __builtin_amdgcn_wmma_f32_16x16x32_f16(
        false, a, false, b, (short)0, c, false, false);
}

// A-matrix fragment (16x32 f16), LDS row stride = 32 halfs.
// lane l: row = l&15 ; elems 0..7 -> K = 8*(l>>4)+e ; elems 8..15 -> 16+8*(l>>4)+e
__device__ __forceinline__ v16h load_a_frag(const _Float16* base, int lane) {
    int r  = lane & 15;
    int kh = (lane >> 4) * 8;
    const _Float16* p = base + r * 32 + kh;
    v16h f;
    *(v8h*)&f       = *(const v8h*)p;         // ds_load_b128
    *((v8h*)&f + 1) = *(const v8h*)(p + 16);
    return f;
}

// B-matrix fragment (32x16 f16) from (n, k) LDS tile, stride 32 halfs.
// lane l: col = l&15 ; elems 0..7 -> K = 8*(l>>4)+e ; elems 8..15 -> 16+8*(l>>4)+e
__device__ __forceinline__ v16h load_b_frag(const _Float16* base, int lane) {
    int c  = lane & 15;
    int kh = (lane >> 4) * 8;
    const _Float16* p = base + c * 32 + kh;
    v16h f;
    *(v8h*)&f       = *(const v8h*)p;
    *((v8h*)&f + 1) = *(const v8h*)(p + 16);
    return f;
}

__device__ __forceinline__ float gelu_tanh(float x) {
    float x3 = x * x * x;
    return 0.5f * x * (1.0f + tanhf(0.7978845608028654f * (x + 0.044715f * x3)));
}

// ---------------------------------------------------------------------------
// weight convert + transpose: dst[l][n][k] = (f16) src[l][k][n]
// ---------------------------------------------------------------------------
__global__ __launch_bounds__(256) void cvt_tr_kernel(
    const float* __restrict__ src, _Float16* __restrict__ dst, int K, int N)
{
    size_t base = (size_t)blockIdx.z * K * N;
    size_t i = (size_t)blockIdx.x * 256 + threadIdx.x;
    if (i >= (size_t)K * N) return;
    int k = (int)(i / N), n = (int)(i % N);
    dst[base + (size_t)n * K + k] = (_Float16)src[base + i];
}

// ---------------------------------------------------------------------------
// im2col (straight to f16): patches[tok][(py*16+px)*3+ci]
// ---------------------------------------------------------------------------
__global__ __launch_bounds__(256) void im2col_kernel(
    const float* __restrict__ image, _Float16* __restrict__ patches)
{
    size_t i = (size_t)blockIdx.x * 256 + threadIdx.x;
    if (i >= (size_t)MTOK * WID) return;
    int col = (int)(i % WID);
    int tok = (int)(i / WID);
    int n  = tok / LTOK;
    int l  = tok % LTOK;
    int gy = l / GRIDP, gx = l % GRIDP;
    int ci = col % 3;
    int px = (col / 3) % 16;
    int py = col / 48;
    patches[i] =
        (_Float16)image[(((size_t)n * 384 + gy * 16 + py) * 384 + gx * 16 + px) * 3 + ci];
}

// ---------------------------------------------------------------------------
// WMMA GEMM: C[M x N] = act(A[M x K] * B[K x N] + bias) (+ residual)
// A: f16 row-major (M x K). Bt: f16 pre-transposed (N x K).
// tile 128x128, 256 thr = 8 waves, wave = 32x64 (2x4 wmma tiles, 8 wmma/step)
// ---------------------------------------------------------------------------
__global__ __launch_bounds__(256) void gemm_kernel(
    const _Float16* __restrict__ A, const _Float16* __restrict__ Bt,
    const float* __restrict__ bias, const float* __restrict__ residual,
    float* __restrict__ Cf, _Float16* __restrict__ Ch,
    int Ndim, int Kdim, int act)
{
    __shared__ _Float16 As[128 * 32];   // 8 KB (m, k)
    __shared__ _Float16 Bs[128 * 32];   // 8 KB (n, k)

    const int tid   = threadIdx.x;
    const int lane  = tid & 31;
    const int wave  = tid >> 5;
    const int wm    = (wave & 3) * 32;    // 0,32,64,96
    const int wn    = (wave >> 2) * 64;   // 0,64
    const int tileM = blockIdx.y * 128;
    const int tileN = blockIdx.x * 128;

    v8f acc[2][4] = {};

    for (int k0 = 0; k0 < Kdim; k0 += 32) {
        // stage both tiles: 512 chunks x 16B each, 4 copies per thread
        for (int c = tid; c < 512; c += 256) {
            int row = c >> 2, ch = (c & 3) * 8;
            copy16_g2l(A  + (size_t)(tileM + row) * Kdim + k0 + ch, &As[row * 32 + ch]);
            copy16_g2l(Bt + (size_t)(tileN + row) * Kdim + k0 + ch, &Bs[row * 32 + ch]);
        }
        wait_copies();
        __syncthreads();

        v16h a0 = load_a_frag(&As[(wm)      * 32], lane);
        v16h a1 = load_a_frag(&As[(wm + 16) * 32], lane);
        v16h b0 = load_b_frag(&Bs[(wn)      * 32], lane);
        v16h b1 = load_b_frag(&Bs[(wn + 16) * 32], lane);
        v16h b2 = load_b_frag(&Bs[(wn + 32) * 32], lane);
        v16h b3 = load_b_frag(&Bs[(wn + 48) * 32], lane);

        acc[0][0] = wmma16(a0, b0, acc[0][0]);
        acc[1][0] = wmma16(a1, b0, acc[1][0]);
        acc[0][1] = wmma16(a0, b1, acc[0][1]);
        acc[1][1] = wmma16(a1, b1, acc[1][1]);
        acc[0][2] = wmma16(a0, b2, acc[0][2]);
        acc[1][2] = wmma16(a1, b2, acc[1][2]);
        acc[0][3] = wmma16(a0, b3, acc[0][3]);
        acc[1][3] = wmma16(a1, b3, acc[1][3]);
        __syncthreads();
    }

    // D layout: VGPR r -> m = r + 8*(lane>>4), n = lane&15
    const int mh = (lane >> 4) * 8;
    const int nc = lane & 15;
    for (int ti = 0; ti < 2; ++ti) {
        for (int tj = 0; tj < 4; ++tj) {
            int cn = tileN + wn + tj * 16 + nc;
            float bvv = bias ? bias[cn] : 0.0f;
            for (int r = 0; r < 8; ++r) {
                int cm = tileM + wm + ti * 16 + mh + r;
                float val = acc[ti][tj][r] + bvv;
                if (act == 1) val = gelu_tanh(val);
                if (residual) val += residual[(size_t)cm * Ndim + cn];
                if (Cf) Cf[(size_t)cm * Ndim + cn] = val;
                if (Ch) Ch[(size_t)cm * Ndim + cn] = (_Float16)val;
            }
        }
    }
}

// ---------------------------------------------------------------------------
// LayerNorm: one block (256 threads) per row; optional f32 and f16 outputs
// ---------------------------------------------------------------------------
__global__ __launch_bounds__(256) void ln_kernel(
    const float* __restrict__ x, const float* __restrict__ scale,
    const float* __restrict__ bias, float* __restrict__ yf,
    _Float16* __restrict__ yh, int Wd)
{
    __shared__ float sred[8];
    const int row = blockIdx.x;
    const float* xr = x + (size_t)row * Wd;
    const int tid = threadIdx.x, lane = tid & 31, wv = tid >> 5;

    float s = 0.0f;
    for (int i = tid; i < Wd; i += 256) s += xr[i];
    for (int o = 16; o; o >>= 1) s += __shfl_down(s, o, 32);
    if (lane == 0) sred[wv] = s;
    __syncthreads();
    float t = (lane < 8) ? sred[lane] : 0.0f;
    for (int o = 4; o; o >>= 1) t += __shfl_down(t, o, 32);
    float mean = __shfl(t, 0, 32) / (float)Wd;
    __syncthreads();

    float vs = 0.0f;
    for (int i = tid; i < Wd; i += 256) { float d = xr[i] - mean; vs += d * d; }
    for (int o = 16; o; o >>= 1) vs += __shfl_down(vs, o, 32);
    if (lane == 0) sred[wv] = vs;
    __syncthreads();
    float tv = (lane < 8) ? sred[lane] : 0.0f;
    for (int o = 4; o; o >>= 1) tv += __shfl_down(tv, o, 32);
    float var = __shfl(tv, 0, 32) / (float)Wd;

    float inv = rsqrtf(var + 1e-6f);
    for (int i = tid; i < Wd; i += 256) {
        float out = (xr[i] - mean) * inv * scale[i] + bias[i];
        if (yf) yf[(size_t)row * Wd + i] = out;
        if (yh) yh[(size_t)row * Wd + i] = (_Float16)out;
    }
}

// ---------------------------------------------------------------------------
// 2D axial RoPE in-place on f16 q and k. Block = token, 384 thr = H*32 pairs.
// ---------------------------------------------------------------------------
__global__ __launch_bounds__(384) void rope_kernel(
    _Float16* __restrict__ q, _Float16* __restrict__ k)
{
    const int tok = blockIdx.x;
    const int l   = tok % LTOK;
    const int gy  = l / GRIDP, gx = l % GRIDP;
    const int t    = threadIdx.x;
    const int h    = t >> 5;
    const int pair = t & 31;                         // 32 pairs per head
    const float pos = (pair < 16) ? ((gx + 0.5f) / (float)GRIDP)
                                  : ((gy + 0.5f) / (float)GRIDP);
    const int f = pair & 15;                         // P = HD/4 = 16 freqs
    const float inv = __powf(10000.0f, -((float)f / 15.0f));
    const float ang = pos * inv;
    float c, s;
    __sincosf(ang, &s, &c);
    const size_t idx = (size_t)tok * WID + h * HDIM + pair * 2;
    float qx = (float)q[idx], qy = (float)q[idx + 1];
    q[idx]     = (_Float16)(qx * c - qy * s);
    q[idx + 1] = (_Float16)(qx * s + qy * c);
    float kx = (float)k[idx], ky = (float)k[idx + 1];
    k[idx]     = (_Float16)(kx * c - ky * s);
    k[idx + 1] = (_Float16)(kx * s + ky * c);
}

// ---------------------------------------------------------------------------
// Attention: 1 wave per (image, head, 16-row Q tile), q/k/v/o in f16.
// S = Q*K^T (WMMA), exact softmax over LDS, O = P*V (WMMA).
// ---------------------------------------------------------------------------
__global__ __launch_bounds__(32) void attn_kernel(
    const _Float16* __restrict__ qg, const _Float16* __restrict__ kg,
    const _Float16* __restrict__ vg, _Float16* __restrict__ og)
{
    __shared__ float S[16 * LTOK];   // 36 KB score rows

    const int lane = threadIdx.x;
    const int q0   = blockIdx.x * 16;
    const int h    = blockIdx.y;
    const int n    = blockIdx.z;
    const float scale = 0.125f;      // 1/sqrt(64)

    const int r    = lane & 15;
    const int kh8  = (lane >> 4) * 8;
    const int kh16 = (lane >> 4) * 16;

    // Q fragments (pre-scaled): d chunks [0,32) and [32,64)
    v16h aq0, aq1;
    {
        const size_t qb = ((size_t)(n * LTOK + q0 + r)) * WID + h * HDIM;
        v8h t0 = *(const v8h*)(qg + qb + kh8);
        v8h t1 = *(const v8h*)(qg + qb + 16 + kh8);
        v8h t2 = *(const v8h*)(qg + qb + 32 + kh8);
        v8h t3 = *(const v8h*)(qg + qb + 48 + kh8);
        for (int e = 0; e < 8; ++e) {
            aq0[e]     = (_Float16)((float)t0[e] * scale);
            aq0[e + 8] = (_Float16)((float)t1[e] * scale);
            aq1[e]     = (_Float16)((float)t2[e] * scale);
            aq1[e + 8] = (_Float16)((float)t3[e] * scale);
        }
    }

    // logits: 36 blocks of 16 keys
    for (int kb = 0; kb < LTOK; kb += 16) {
        const size_t kr = ((size_t)(n * LTOK + kb + r)) * WID + h * HDIM;
        v16h bk0, bk1;   // B = K^T, col = key, table-order d runs
        *(v8h*)&bk0       = *(const v8h*)(kg + kr + kh8);
        *((v8h*)&bk0 + 1) = *(const v8h*)(kg + kr + kh8 + 16);
        *(v8h*)&bk1       = *(const v8h*)(kg + kr + 32 + kh8);
        *((v8h*)&bk1 + 1) = *(const v8h*)(kg + kr + 32 + kh8 + 16);
        v8f s = {};
        s = wmma16(aq0, bk0, s);
        s = wmma16(aq1, bk1, s);
        const int jj = kb + (lane & 15);
        const int mb = (lane >> 4) * 8;
        for (int rr = 0; rr < 8; ++rr) S[(mb + rr) * LTOK + jj] = s[rr];
    }
    __syncthreads();

    // exact softmax, one lane per query row
    if (lane < 16) {
        float* sr = &S[lane * LTOK];
        float mx = -1e30f;
        for (int j = 0; j < LTOK; ++j) mx = fmaxf(mx, sr[j]);
        float sum = 0.0f;
        for (int j = 0; j < LTOK; ++j) { float e = __expf(sr[j] - mx); sr[j] = e; sum += e; }
        float invs = 1.0f / sum;
        for (int j = 0; j < LTOK; ++j) sr[j] *= invs;
    }
    __syncthreads();

    // O(16x64) = P(16x576) * V(576x64), 32-key blocks, 4 col chunks
    v8f oacc[4] = {};
    for (int kb = 0; kb < LTOK; kb += 32) {
        v16h ap;   // P fragment (16x32) from LDS (f32 -> f16), A layout
        for (int e = 0; e < 8; ++e) {
            ap[e]     = (_Float16)S[r * LTOK + kb + kh16 + e];
            ap[e + 8] = (_Float16)S[r * LTOK + kb + kh16 + 8 + e];
        }
        for (int j = 0; j < 4; ++j) {
            v16h bvf;  // V chunk (32x16): col = j*16+(lane&15), row = key
            const size_t vb = ((size_t)(n * LTOK + kb + kh16)) * WID + h * HDIM + j * 16 + r;
            for (int e = 0; e < 16; ++e)
                bvf[e] = vg[vb + (size_t)e * WID];
            oacc[j] = wmma16(ap, bvf, oacc[j]);
        }
    }

    const int cn = lane & 15;
    const int mb = (lane >> 4) * 8;
    for (int j = 0; j < 4; ++j)
        for (int rr = 0; rr < 8; ++rr) {
            int m = q0 + mb + rr;
            og[((size_t)(n * LTOK + m)) * WID + h * HDIM + j * 16 + cn] =
                (_Float16)oacc[j][rr];
        }
}

// ---------------------------------------------------------------------------
// final mean pool over tokens: out[n][c] = mean_l y[n,l,c]
// ---------------------------------------------------------------------------
__global__ __launch_bounds__(256) void mean_kernel(
    const float* __restrict__ y, float* __restrict__ out)
{
    int i = blockIdx.x * 256 + threadIdx.x;
    if (i >= NIMG * WID) return;
    int n = i / WID, c = i % WID;
    float s = 0.0f;
    for (int l = 0; l < LTOK; ++l)
        s += y[((size_t)(n * LTOK + l)) * WID + c];
    out[i] = s * (1.0f / (float)LTOK);
}

// ---------------------------------------------------------------------------
extern "C" void kernel_launch(void* const* d_in, const int* in_sizes, int n_in,
                              void* d_out, int out_size, void* d_ws, size_t ws_size,
                              hipStream_t stream)
{
    (void)in_sizes; (void)n_in; (void)out_size; (void)ws_size;

    const float* image = (const float*)d_in[0];
    const float* convW = (const float*)d_in[1];
    const float* convB = (const float*)d_in[2];
    const float* ln1s  = (const float*)d_in[3];
    const float* ln1b  = (const float*)d_in[4];
    const float* wq    = (const float*)d_in[5];
    const float* bq    = (const float*)d_in[6];
    const float* wk    = (const float*)d_in[7];
    const float* bk    = (const float*)d_in[8];
    const float* wv    = (const float*)d_in[9];
    const float* bv    = (const float*)d_in[10];
    const float* wo    = (const float*)d_in[11];
    const float* bo    = (const float*)d_in[12];
    const float* ln2s  = (const float*)d_in[13];
    const float* ln2b  = (const float*)d_in[14];
    const float* w1    = (const float*)d_in[15];
    const float* b1    = (const float*)d_in[16];
    const float* w2    = (const float*)d_in[17];
    const float* b2    = (const float*)d_in[18];
    const float* lnfs  = (const float*)d_in[19];
    const float* lnfb  = (const float*)d_in[20];

    // ---- workspace layout ----
    const size_t MW  = (size_t)MTOK * WID;      // 3.54M elems
    const size_t MH  = (size_t)MTOK * MLPD;     // 14.16M elems
    const size_t WW  = (size_t)WID * WID;       // 589824
    const size_t WM  = (size_t)WID * MLPD;      // 2.36M

    float* f32p = (float*)d_ws;
    float* x = f32p;           f32p += MW;      // residual stream
    float* y = f32p;           f32p += MW;      // final LN output

    _Float16* f16p = (_Float16*)f32p;
    _Float16* yh      = f16p;  f16p += MW;
    _Float16* qh      = f16p;  f16p += MW;
    _Float16* kh      = f16p;  f16p += MW;
    _Float16* vh      = f16p;  f16p += MW;
    _Float16* oh      = f16p;  f16p += MW;
    _Float16* hidh    = f16p;  f16p += MH;
    _Float16* patch_h = f16p;  f16p += MW;
    _Float16* convW_h = f16p;  f16p += WW;
    _Float16* wq_h    = f16p;  f16p += DEPTH * WW;
    _Float16* wk_h    = f16p;  f16p += DEPTH * WW;
    _Float16* wv_h    = f16p;  f16p += DEPTH * WW;
    _Float16* wo_h    = f16p;  f16p += DEPTH * WW;
    _Float16* w1_h    = f16p;  f16p += DEPTH * WM;
    _Float16* w2_h    = f16p;  f16p += DEPTH * WM;

    const dim3 blk256(256);
    const unsigned bWW = (unsigned)((WW + 255) / 256);
    const unsigned bWM = (unsigned)((WM + 255) / 256);

    // ---- weight convert + transpose (f32 KxN -> f16 NxK) ----
    cvt_tr_kernel<<<dim3(bWW, 1, 1),     blk256, 0, stream>>>(convW, convW_h, WID, WID);
    cvt_tr_kernel<<<dim3(bWW, 1, DEPTH), blk256, 0, stream>>>(wq, wq_h, WID, WID);
    cvt_tr_kernel<<<dim3(bWW, 1, DEPTH), blk256, 0, stream>>>(wk, wk_h, WID, WID);
    cvt_tr_kernel<<<dim3(bWW, 1, DEPTH), blk256, 0, stream>>>(wv, wv_h, WID, WID);
    cvt_tr_kernel<<<dim3(bWW, 1, DEPTH), blk256, 0, stream>>>(wo, wo_h, WID, WID);
    cvt_tr_kernel<<<dim3(bWM, 1, DEPTH), blk256, 0, stream>>>(w1, w1_h, WID, MLPD);
    cvt_tr_kernel<<<dim3(bWM, 1, DEPTH), blk256, 0, stream>>>(w2, w2_h, MLPD, WID);

    const dim3 gemm768(WID / 128, MTOK / 128);   // (6, 36)
    const dim3 gemmMLP(MLPD / 128, MTOK / 128);  // (24, 36)

    // ---- patch embed ----
    im2col_kernel<<<dim3((unsigned)((MW + 255) / 256)), blk256, 0, stream>>>(image, patch_h);
    gemm_kernel<<<gemm768, blk256, 0, stream>>>(patch_h, convW_h, convB, nullptr,
                                                x, nullptr, WID, WID, 0);

    for (int l = 0; l < DEPTH; ++l) {
        const size_t wOff  = (size_t)l * WW;
        const size_t mOff  = (size_t)l * WM;

        ln_kernel<<<MTOK, blk256, 0, stream>>>(x, ln1s + l * WID, ln1b + l * WID,
                                               nullptr, yh, WID);

        gemm_kernel<<<gemm768, blk256, 0, stream>>>(yh, wq_h + wOff, bq + l * WID, nullptr,
                                                    nullptr, qh, WID, WID, 0);
        gemm_kernel<<<gemm768, blk256, 0, stream>>>(yh, wk_h + wOff, bk + l * WID, nullptr,
                                                    nullptr, kh, WID, WID, 0);
        gemm_kernel<<<gemm768, blk256, 0, stream>>>(yh, wv_h + wOff, bv + l * WID, nullptr,
                                                    nullptr, vh, WID, WID, 0);

        rope_kernel<<<MTOK, dim3(NHEAD * 32), 0, stream>>>(qh, kh);

        attn_kernel<<<dim3(LTOK / 16, NHEAD, NIMG), dim3(32), 0, stream>>>(qh, kh, vh, oh);

        gemm_kernel<<<gemm768, blk256, 0, stream>>>(oh, wo_h + wOff, bo + l * WID, x,
                                                    x, nullptr, WID, WID, 0);

        ln_kernel<<<MTOK, blk256, 0, stream>>>(x, ln2s + l * WID, ln2b + l * WID,
                                               nullptr, yh, WID);

        gemm_kernel<<<gemmMLP, blk256, 0, stream>>>(yh, w1_h + mOff, b1 + l * MLPD, nullptr,
                                                    nullptr, hidh, MLPD, WID, 1 /*GELU*/);
        gemm_kernel<<<gemm768, blk256, 0, stream>>>(hidh, w2_h + mOff, b2 + l * WID, x,
                                                    x, nullptr, WID, MLPD, 0);
    }

    ln_kernel<<<MTOK, blk256, 0, stream>>>(x, lnfs, lnfb, y, nullptr, WID);
    mean_kernel<<<dim3((NIMG * WID + 255) / 256), blk256, 0, stream>>>(y, (float*)d_out);
}
